// VNResnetPointnet_62723702391736
// MI455X (gfx1250) — matrix operand order repaired and checked by reading, not compiled, not run on hardware
//
#include <hip/hip_runtime.h>

// ---------------------------------------------------------------------------
// VN-ResNet PointNet encoder for MI455X (gfx1250), wave32 + WMMA bf16.
//   1) knn_mean:     mean of k=20 nearest neighbors per point (LDS-staged)
//   2) feat_fcpos:   feat_mean = {m-p, p, cross(m,p)}; X0 = fc_pos_W @ feat
//   3) 5 x block:    GEMM(d0) -> VN-ReLU -> GEMM(W0) -> GEMM(d1) -> VN-ReLU
//                    -> GEMM(Ws)->S ; GEMM(W1, C=S) -> NET       (all WMMA)
//   4) pool + head
// GEMMs stage the shared 64-column B tile in LDS via
// global_load_async_to_lds_b128 (ASYNCcnt) so all 8 waves of a workgroup
// reuse it; A (weights) streams through WGP$/L2 with global_prefetch_b8.
// Activation layout: [NC=24576 rows (b,v,n)-major][ST=256 channels] bf16.
// ---------------------------------------------------------------------------

typedef __attribute__((ext_vector_type(8)))  float  v8f;
typedef __attribute__((ext_vector_type(8)))  __bf16 v8bf;
typedef __attribute__((ext_vector_type(16))) __bf16 v16bf;

#define BATCH  4
#define NPTS   2048
#define NPOINT (BATCH * NPTS)       // 8192
#define NC     (BATCH * 3 * NPTS)   // 24576 GEMM columns
#define ST     256                  // channel stride of activation buffers
#define KNN    20

// ------------------------------ f32 -> bf16 --------------------------------
__global__ void f2bf_kernel(const float* __restrict__ s, __bf16* __restrict__ d, int n) {
    int i = blockIdx.x * 256 + threadIdx.x;
    if (i < n) d[i] = (__bf16)s[i];
}

// ------------------------------ kNN mean -----------------------------------
// grid = 32 blocks (8 per batch), 256 threads; each thread = one query point.
__global__ __launch_bounds__(256)
void knn_mean_kernel(const float* __restrict__ p, float* __restrict__ nbr_mean) {
    __shared__ float sx[NPTS], sy[NPTS], sz[NPTS];
    const int b     = blockIdx.x >> 3;
    const int qbase = (blockIdx.x & 7) * 256;
    const float* pb = p + (size_t)b * NPTS * 3;
    for (int i = threadIdx.x; i < NPTS; i += 256) {
        sx[i] = pb[i * 3 + 0];
        sy[i] = pb[i * 3 + 1];
        sz[i] = pb[i * 3 + 2];
    }
    __syncthreads();
    const int q = qbase + threadIdx.x;
    const float px = sx[q], py = sy[q], pz = sz[q];
    float bd[KNN]; int bi[KNN];
#pragma unroll
    for (int t = 0; t < KNN; ++t) { bd[t] = -3.4e38f; bi[t] = 0; }
    for (int j = 0; j < NPTS; ++j) {
        const float dx = px - sx[j], dy = py - sy[j], dz = pz - sz[j];
        const float v = -(dx * dx + dy * dy + dz * dz);   // neg squared dist
        if (v > bd[0]) {                                  // beats current worst
            int pos = 0;
            while (pos < KNN - 1 && bd[pos + 1] < v) {
                bd[pos] = bd[pos + 1]; bi[pos] = bi[pos + 1]; ++pos;
            }
            bd[pos] = v; bi[pos] = j;
        }
    }
    float mx = 0.f, my = 0.f, mz = 0.f;
#pragma unroll
    for (int t = 0; t < KNN; ++t) { int j = bi[t]; mx += sx[j]; my += sy[j]; mz += sz[j]; }
    const float inv = 1.0f / (float)KNN;
    const size_t o = ((size_t)b * NPTS + q) * 3;
    nbr_mean[o + 0] = mx * inv;
    nbr_mean[o + 1] = my * inv;
    nbr_mean[o + 2] = mz * inv;
}

// --------------------- feat_mean + fc_pos -> X0 (bf16) ---------------------
// grid = NC blocks, 256 threads (one per output channel).
__global__ __launch_bounds__(256)
void feat_fcpos_kernel(const float* __restrict__ p, const float* __restrict__ nm,
                       const float* __restrict__ Wfc, __bf16* __restrict__ X) {
    const int c = blockIdx.x;          // row = (b*3+v)*N + n
    const int o = threadIdx.x;
    const int b = c / (3 * NPTS);
    const int v = (c / NPTS) % 3;
    const int n = c & (NPTS - 1);
    const size_t pi = ((size_t)b * NPTS + n) * 3;
    const float p0 = p[pi], p1 = p[pi + 1], p2 = p[pi + 2];
    const float m0 = nm[pi], m1 = nm[pi + 1], m2 = nm[pi + 2];
    // cross(m, p)
    const float cx = m1 * p2 - m2 * p1;
    const float cy = m2 * p0 - m0 * p2;
    const float cz = m0 * p1 - m1 * p0;
    const float pv = (v == 0) ? p0 : ((v == 1) ? p1 : p2);
    const float mv = (v == 0) ? m0 : ((v == 1) ? m1 : m2);
    const float cv = (v == 0) ? cx : ((v == 1) ? cy : cz);
    const float f0 = mv - pv, f1 = pv, f2 = cv;
    const float y = Wfc[o * 3 + 0] * f0 + Wfc[o * 3 + 1] * f1 + Wfc[o * 3 + 2] * f2;
    X[(size_t)c * ST + o] = (__bf16)y;
}

// ------------------------------ WMMA GEMM ----------------------------------
// Y[M, NC] = W[M,K] @ X[K, NC]; X stored row=column-index, K contiguous,
// global row stride ST. Workgroup = 8 waves = 8 M-tiles x one 64-col group.
// B tile (64 x K bf16, <=32KB) staged in LDS via async global->LDS copies
// (ASYNCcnt), consumed by all 8 waves. Optional f32 C-init (fused shortcut);
// output bf16 or f32.
__global__ __launch_bounds__(256)
void gemm_wmma_kernel(const __bf16* __restrict__ W, const __bf16* __restrict__ X,
                      int M, int K,
                      const float* __restrict__ Cin, int Cstride,
                      __bf16* __restrict__ Yb, float* __restrict__ Yf, int Ostride) {
    __shared__ __attribute__((aligned(16))) __bf16 smem[64 * 256];  // 32KB max

    const int tid  = threadIdx.x;
    const int lane = tid & 31;
    const int wave = tid >> 5;
    const int WGperM = (M >> 4) >> 3;          // 1 (M=128) or 2 (M=256)
    const int ng = blockIdx.x / WGperM;        // 64-column group index
    const int mw = blockIdx.x % WGperM;        // which M half
    const int m0 = (mw * 8 + wave) * 16;
    const int n0 = ng * 64;
    const int l15 = lane & 15;
    const int h   = lane >> 4;

    // ---- async stage X[n0..n0+63][0..K) into LDS (packed rows of K) ----
    // LDS byte address = low 32 bits of the generic pointer (LDS aperture
    // maps addr[31:0] -> LDS offset). Each lane copies 16B per issue;
    // totalChunks is a multiple of 256 so EXEC stays all-ones.
    const int rowBytes     = K * 2;            // 256 or 512
    const int chunksPerRow = rowBytes >> 4;    // 16 or 32
    const int totalChunks  = 64 * chunksPerRow;
    const unsigned ldsBase = (unsigned)(unsigned long long)(const void*)&smem[0];
    for (int c = tid; c < totalChunks; c += 256) {
        const int row = c / chunksPerRow;
        const int col = (c - row * chunksPerRow) * 16;   // byte offset in row
        const unsigned long long ga =
            (unsigned long long)(const void*)((const char*)(X + (size_t)(n0 + row) * ST) + col);
        const unsigned la = ldsBase + (unsigned)(row * rowBytes + col);
        asm volatile("global_load_async_to_lds_b128 %0, %1, off"
                     :: "v"(la), "v"(ga) : "memory");
    }
    asm volatile("s_wait_asynccnt 0x0" ::: "memory");
    __syncthreads();

    v8f acc[4];
    if (Cin) {
#pragma unroll
        for (int j = 0; j < 4; ++j)
            acc[j] = *(const v8f*)(Cin + (size_t)(n0 + j * 16 + l15) * Cstride + m0 + 8 * h);
    } else {
#pragma unroll
        for (int j = 0; j < 4; ++j)
            acc[j] = (v8f){0.f, 0.f, 0.f, 0.f, 0.f, 0.f, 0.f, 0.f};
    }

    // A fragment: lane-half h holds K = h*16..h*16+15 of row m.
    const __bf16* aBase = W + (size_t)(m0 + l15) * K + h * 16;
    // B fragment from LDS: lanes 0-15 K=0..15, lanes 16-31 K=16..31 of col n.
    const __bf16* bBase = smem + (size_t)l15 * K + h * 16;

    for (int k0 = 0; k0 < K; k0 += 32) {
        __builtin_prefetch(aBase + k0 + 32, 0, 0);
        union { v16bf v; v8bf half[2]; } a;
        a.half[0] = *(const v8bf*)(aBase + k0);
        a.half[1] = *(const v8bf*)(aBase + k0 + 8);
#pragma unroll
        for (int j = 0; j < 4; ++j) {
            v16bf bfrag = *(const v16bf*)(bBase + (size_t)j * 16 * K + k0);
            acc[j] = __builtin_amdgcn_wmma_f32_16x16x32_bf16(
                false, a.v, false, bfrag, (short)0, acc[j], false, false);
        }
    }

#pragma unroll
    for (int j = 0; j < 4; ++j) {
        const size_t row = (size_t)(n0 + j * 16 + l15);
        if (Yf) {
            *(v8f*)(Yf + row * Ostride + m0 + 8 * h) = acc[j];
        } else {
            v8bf o;
#pragma unroll
            for (int e = 0; e < 8; ++e) o[e] = (__bf16)acc[j][e];
            *(v8bf*)(Yb + row * Ostride + m0 + 8 * h) = o;
        }
    }
}

// ------------------------------ VN-ReLU ------------------------------------
// out = (dot>=0) ? x : x - (dot/dsq)*d, dot/dsq over the 3-vector axis.
// grid = NPOINT blocks, nch threads.
__global__ void vn_relu_kernel(const __bf16* __restrict__ X, const __bf16* __restrict__ D,
                               __bf16* __restrict__ O) {
    const int pid = blockIdx.x;
    const int o   = threadIdx.x;
    const int b = pid >> 11;
    const int n = pid & (NPTS - 1);
    float x[3], d[3];
    size_t rows[3];
#pragma unroll
    for (int v = 0; v < 3; ++v) {
        rows[v] = ((size_t)(b * 3 + v) * NPTS + n) * ST + o;
        x[v] = (float)X[rows[v]];
        d[v] = (float)D[rows[v]];
    }
    const float dot = x[0] * d[0] + x[1] * d[1] + x[2] * d[2];
    const float dsq = d[0] * d[0] + d[1] * d[1] + d[2] * d[2] + 1e-8f;
    const float s = (dot >= 0.f) ? 0.f : (dot / dsq);
#pragma unroll
    for (int v = 0; v < 3; ++v) O[rows[v]] = (__bf16)(x[v] - s * d[v]);
}

// --------------------- mean over n per (b,v) group -------------------------
// grid = 12 blocks, 128 threads (one per channel).
__global__ __launch_bounds__(128)
void pool_mean_kernel(const __bf16* __restrict__ NET, float* __restrict__ pooled) {
    const int g  = blockIdx.x;      // g = b*3+v
    const int ch = threadIdx.x;
    const __bf16* base = NET + (size_t)g * NPTS * ST + ch;
    float s = 0.f;
    for (int n = 0; n < NPTS; ++n) s += (float)base[(size_t)n * ST];
    pooled[g * 128 + ch] = s * (1.0f / (float)NPTS);
}

// ----------- X = concat(NET, broadcast(pooled)) along channels -------------
__global__ __launch_bounds__(256)
void build_xin_kernel(const __bf16* __restrict__ NET, const float* __restrict__ pooled,
                      __bf16* __restrict__ X) {
    const size_t c = blockIdx.x;
    const int o = threadIdx.x;
    const int g = (int)(c / NPTS);
    __bf16 val = (o < 128) ? NET[c * ST + o] : (__bf16)pooled[g * 128 + (o - 128)];
    X[c * ST + o] = val;
}

// ------------------------------ final head ---------------------------------
// c = fc_c_W @ vn_relu(act_d, net_mean); out[b, o*3+v]. One block, 128 thr.
__global__ __launch_bounds__(128)
void final_head_kernel(const float* __restrict__ netm /* [12][128] */,
                       const float* __restrict__ act_d, const float* __restrict__ fc_c,
                       float* __restrict__ out) {
    __shared__ float y[3][128];
    const int o = threadIdx.x;
    for (int b = 0; b < BATCH; ++b) {
        float d0 = 0.f, d1 = 0.f, d2 = 0.f;
        for (int i = 0; i < 128; ++i) {
            const float w = act_d[o * 128 + i];
            d0 += w * netm[(b * 3 + 0) * 128 + i];
            d1 += w * netm[(b * 3 + 1) * 128 + i];
            d2 += w * netm[(b * 3 + 2) * 128 + i];
        }
        const float x0 = netm[(b * 3 + 0) * 128 + o];
        const float x1 = netm[(b * 3 + 1) * 128 + o];
        const float x2 = netm[(b * 3 + 2) * 128 + o];
        const float dot = x0 * d0 + x1 * d1 + x2 * d2;
        const float dsq = d0 * d0 + d1 * d1 + d2 * d2 + 1e-8f;
        const float s = (dot >= 0.f) ? 0.f : (dot / dsq);
        __syncthreads();
        y[0][o] = x0 - s * d0;
        y[1][o] = x1 - s * d1;
        y[2][o] = x2 - s * d2;
        __syncthreads();
        float c0 = 0.f, c1 = 0.f, c2 = 0.f;
        for (int i = 0; i < 128; ++i) {
            const float w = fc_c[o * 128 + i];
            c0 += w * y[0][i];
            c1 += w * y[1][i];
            c2 += w * y[2][i];
        }
        out[b * 384 + o * 3 + 0] = c0;
        out[b * 384 + o * 3 + 1] = c1;
        out[b * 384 + o * 3 + 2] = c2;
    }
}

// ------------------------------- launcher ----------------------------------
static inline int gemm_blocks(int M) { return ((M == 256) ? 2 : 1) * (NC / 64); }

extern "C" void kernel_launch(void* const* d_in, const int* in_sizes, int n_in,
                              void* d_out, int out_size, void* d_ws, size_t ws_size,
                              hipStream_t stream) {
    const float* p      = (const float*)d_in[0];  // [4,2048,3]
    const float* fc_pos = (const float*)d_in[1];  // [256,3]
    const float* blk_d0 = (const float*)d_in[2];  // [5,256,256]
    const float* blk_W0 = (const float*)d_in[3];  // [5,128,256]
    const float* blk_d1 = (const float*)d_in[4];  // [5,128,128]
    const float* blk_W1 = (const float*)d_in[5];  // [5,128,128]
    const float* blk_Ws = (const float*)d_in[6];  // [5,128,256]
    const float* fc_c   = (const float*)d_in[7];  // [128,128]
    const float* act_d  = (const float*)d_in[8];  // [128,128]
    float* out = (float*)d_out;                   // [4,384]

    // ---- workspace carve-up (aligned to 256B) ----
    char* ws = (char*)d_ws;
    size_t off = 0;
    auto carve = [&](size_t bytes) -> void* {
        void* ptr = ws + off;
        off += (bytes + 255) & ~(size_t)255;
        return ptr;
    };
    __bf16* d0b = (__bf16*)carve((size_t)5 * 256 * 256 * 2);
    __bf16* W0b = (__bf16*)carve((size_t)5 * 128 * 256 * 2);
    __bf16* d1b = (__bf16*)carve((size_t)5 * 128 * 128 * 2);
    __bf16* W1b = (__bf16*)carve((size_t)5 * 128 * 128 * 2);
    __bf16* Wsb = (__bf16*)carve((size_t)5 * 128 * 256 * 2);
    __bf16* X   = (__bf16*)carve((size_t)NC * ST * 2);       // block input (256 ch)
    __bf16* T1  = (__bf16*)carve((size_t)NC * ST * 2);       // scratch
    __bf16* T2  = (__bf16*)carve((size_t)NC * ST * 2);       // scratch
    __bf16* NET = (__bf16*)carve((size_t)NC * ST * 2);       // block output (128 ch)
    float*  S   = (float*) carve((size_t)NC * 128 * 4);      // shortcut, f32
    float*  pooled = (float*)carve((size_t)12 * 128 * 4);
    float*  nbrm   = (float*)carve((size_t)NPOINT * 3 * 4);
    (void)ws_size; (void)in_sizes; (void)n_in; (void)out_size;

    // ---- convert weights to bf16 once per launch ----
    auto conv = [&](const float* s, __bf16* d, int n) {
        f2bf_kernel<<<(n + 255) / 256, 256, 0, stream>>>(s, d, n);
    };
    conv(blk_d0, d0b, 5 * 256 * 256);
    conv(blk_W0, W0b, 5 * 128 * 256);
    conv(blk_d1, d1b, 5 * 128 * 128);
    conv(blk_W1, W1b, 5 * 128 * 128);
    conv(blk_Ws, Wsb, 5 * 128 * 256);

    // ---- stage 1: kNN mean + feature lift ----
    knn_mean_kernel<<<32, 256, 0, stream>>>(p, nbrm);
    feat_fcpos_kernel<<<NC, 256, 0, stream>>>(p, nbrm, fc_pos, X);

    // ---- stage 2: 5 ResNet blocks ----
    for (int i = 0; i < 5; ++i) {
        if (i > 0) {
            pool_mean_kernel<<<12, 128, 0, stream>>>(NET, pooled);
            build_xin_kernel<<<NC, 256, 0, stream>>>(NET, pooled, X);
        }
        // d = d0 @ X   (256x256)
        gemm_wmma_kernel<<<gemm_blocks(256), 256, 0, stream>>>(
            d0b + (size_t)i * 256 * 256, X, 256, 256, nullptr, 0, T1, nullptr, ST);
        // r1 = vn_relu(X, d)  (256 ch)
        vn_relu_kernel<<<NPOINT, 256, 0, stream>>>(X, T1, T2);
        // h = W0 @ r1  (128x256)
        gemm_wmma_kernel<<<gemm_blocks(128), 256, 0, stream>>>(
            W0b + (size_t)i * 128 * 256, T2, 128, 256, nullptr, 0, T1, nullptr, ST);
        // dh = d1 @ h  (128x128)
        gemm_wmma_kernel<<<gemm_blocks(128), 256, 0, stream>>>(
            d1b + (size_t)i * 128 * 128, T1, 128, 128, nullptr, 0, T2, nullptr, ST);
        // r2 = vn_relu(h, dh)  (128 ch, in-place on T2)
        vn_relu_kernel<<<NPOINT, 128, 0, stream>>>(T1, T2, T2);
        // S = Ws @ X   (128x256, f32 shortcut)
        gemm_wmma_kernel<<<gemm_blocks(128), 256, 0, stream>>>(
            Wsb + (size_t)i * 128 * 256, X, 128, 256, nullptr, 0, nullptr, S, 128);
        // NET = W1 @ r2 + S   (128x128, fused add via WMMA C operand)
        gemm_wmma_kernel<<<gemm_blocks(128), 256, 0, stream>>>(
            W1b + (size_t)i * 128 * 128, T2, 128, 128, S, 128, NET, nullptr, ST);
    }

    // ---- stage 3: global mean + head ----
    pool_mean_kernel<<<12, 128, 0, stream>>>(NET, pooled);
    final_head_kernel<<<1, 128, 0, stream>>>(pooled, act_d, fc_c, out);
}
